// mLSTMfwbw_68384469287276
// MI455X (gfx1250) — matrix operand order repaired and checked by reading, not compiled, not run on hardware
//
#include <hip/hip_runtime.h>

#define TSEQ 4096
#define DK 128
#define DV 128
#define SCH 128   // chunk size
#define NCH 32    // chunks
#define NBH 16    // B*H
#define SB 132    // u16 LDS row stride (bank-conflict padding)

typedef __attribute__((ext_vector_type(8)))  float  v8f;
typedef __attribute__((ext_vector_type(16))) __bf16 v16bf;

static __device__ __forceinline__ unsigned short f2bfu(float f) {
  unsigned int u = __builtin_bit_cast(unsigned int, f);
  unsigned int r = u + 0x7FFFu + ((u >> 16) & 1u);   // round-to-nearest-even
  return (unsigned short)(r >> 16);
}
static __device__ __forceinline__ __bf16 us2bf(unsigned short s) {
  return __builtin_bit_cast(__bf16, s);
}
static __device__ __forceinline__ float bfu2f(unsigned short s) {
  return __builtin_bit_cast(float, ((unsigned int)s) << 16);
}

// Load A operand tile (16x32, rows s0..s0+15, K window kb16..kb16+31) from a
// row-major bf16 LDS matrix with stride SB. Two contiguous 8xu16 runs per lane
// -> lowers to ds_load_b128 pairs.
static __device__ __forceinline__ v16bf load_a(const unsigned short* base,
                                               int s0, int mr, int hh, int kb16) {
  const unsigned short* ar = base + (s0 + mr) * SB + kb16 + 16 * hh;
  v16bf A;
  #pragma unroll
  for (int e = 0; e < 8; ++e) { A[e] = us2bf(ar[e]); A[e + 8] = us2bf(ar[8 + e]); }
  return A;
}
// Load B operand tile (32x16, cols ct*16.., K window kb16) from an [n][K]
// bf16 LDS matrix with stride SB. 16 contiguous u16 per lane -> ds_load_b128.
static __device__ __forceinline__ v16bf load_b(const unsigned short* base,
                                               int ct, int mr, int hh, int kb16) {
  const unsigned short* br = base + (ct * 16 + mr) * SB + kb16 + 16 * hh;
  v16bf B;
  #pragma unroll
  for (int e = 0; e < 16; ++e) B[e] = us2bf(br[e]);
  return B;
}

// ---------------------------------------------------------------------------
// Pass 1: per (head,chunk) gates + kv = (k*fgexp)^T v  (WMMA) + ksum
// ---------------------------------------------------------------------------
__global__ __launch_bounds__(256) void mlstm_p1(
    const float* __restrict__ kg, const float* __restrict__ vg,
    const float* __restrict__ ig, const float* __restrict__ fg,
    float* __restrict__ ws_kv, float* __restrict__ ws_ksum,
    float* __restrict__ ws_lfacc, float* __restrict__ ws_pm,
    float* __restrict__ ws_lfl, float* __restrict__ ws_mi)
{
  extern __shared__ char smem[];
  unsigned short* KB  = (unsigned short*)smem;    // k chunk, [t][k], bf16
  unsigned short* VT  = KB + SCH * SB;            // v chunk, [v][t], bf16
  unsigned short* KFT = VT + SCH * SB;            // (k*fgexp)^T, [k][t], bf16
  float* lfa = (float*)(KFT + SCH * SB);
  float* ivv = lfa + SCH;
  float* pmv = ivv + SCH;                         // lf, then prefix-max
  float* fgx = pmv + SCH;
  float* scl = fgx + SCH;

  const int tid = threadIdx.x;
  const int blk = blockIdx.x;            // bh*32 + chunk
  const int bh  = blk >> 5;
  const int j   = blk & 31;
  const int w = tid >> 5, ln = tid & 31, mr = ln & 15, hh = ln >> 4;
  const int s0 = w * 16;

  const float* kc = kg + ((size_t)bh * TSEQ + (size_t)j * SCH) * DK;
  const float* vc = vg + ((size_t)bh * TSEQ + (size_t)j * SCH) * DV;
  const float* ic = ig + (size_t)bh * TSEQ + (size_t)j * SCH;
  const float* fc = fg + (size_t)bh * TSEQ + (size_t)j * SCH;

  for (int e = tid; e < SCH * DK / 4; e += 256) {
    int row = e >> 5;
    int c4  = (e & 31) << 2;
    float4 xk = *reinterpret_cast<const float4*>(kc + row * DK + c4);
    KB[row * SB + c4 + 0] = f2bfu(xk.x);
    KB[row * SB + c4 + 1] = f2bfu(xk.y);
    KB[row * SB + c4 + 2] = f2bfu(xk.z);
    KB[row * SB + c4 + 3] = f2bfu(xk.w);
    float4 xv = *reinterpret_cast<const float4*>(vc + row * DV + c4);
    VT[(c4 + 0) * SB + row] = f2bfu(xv.x);
    VT[(c4 + 1) * SB + row] = f2bfu(xv.y);
    VT[(c4 + 2) * SB + row] = f2bfu(xv.z);
    VT[(c4 + 3) * SB + row] = f2bfu(xv.w);
  }
  if (tid < SCH) {
    ivv[tid] = ic[tid];
    float fv = fc[tid];
    pmv[tid] = fminf(fv, 0.f) - log1pf(expf(-fabsf(fv)));   // log_sigmoid(f)
  }
  __syncthreads();
  if (tid == 0) {
    float a = 0.f;
    for (int t = 0; t < SCH; ++t) { a += pmv[t]; lfa[t] = a; }   // cumsum
    float lfl = a;
    float pm = -1e30f;
    for (int t = 0; t < SCH; ++t) { pm = fmaxf(pm, ivv[t] - lfa[t]); pmv[t] = pm; }
    scl[0] = lfl;
    scl[1] = lfl + pm;                    // m_intra
    ws_lfl[blk] = lfl;
    ws_mi[blk]  = lfl + pm;
  }
  __syncthreads();
  if (tid < SCH) {
    fgx[tid] = expf(ivv[tid] + scl[0] - lfa[tid] - scl[1]);
    ws_lfacc[(size_t)blk * SCH + tid] = lfa[tid];
    ws_pm[(size_t)blk * SCH + tid]    = pmv[tid];
  }
  __syncthreads();
  // Materialize KFT[k][t] = k[t][k] * fgexp[t]  (transposed, weighted, bf16)
  for (int e = tid; e < SCH * DK; e += 256) {
    int t  = e >> 7;
    int kk = e & 127;
    KFT[kk * SB + t] = f2bfu(bfu2f(KB[t * SB + kk]) * fgx[t]);
  }
  __syncthreads();
  if (tid < SCH) {
    float ks = 0.f;
    for (int t = 0; t < SCH; ++t) ks += bfu2f(KFT[tid * SB + t]);
    ws_ksum[(size_t)blk * DK + tid] = ks;
  }

  // kv[k][v] = sum_t KFT[k][t] * v[t][v]
  v8f acc[8];
  v8f z8 = {0.f, 0.f, 0.f, 0.f, 0.f, 0.f, 0.f, 0.f};
  #pragma unroll
  for (int ct = 0; ct < 8; ++ct) acc[ct] = z8;
  #pragma unroll
  for (int kt = 0; kt < 4; ++kt) {
    const int kb16 = kt * 32;
    v16bf A = load_a(KFT, s0, mr, hh, kb16);
    #pragma unroll
    for (int ct = 0; ct < 8; ++ct) {
      v16bf Bv = load_b(VT, ct, mr, hh, kb16);
      acc[ct] = __builtin_amdgcn_wmma_f32_16x16x32_bf16(
          false, A, false, Bv, (short)0, acc[ct], false, false);
    }
  }
  float* kvout = ws_kv + (size_t)blk * (DK * DV);
  #pragma unroll
  for (int ct = 0; ct < 8; ++ct) {
    #pragma unroll
    for (int r = 0; r < 8; ++r) {
      int kdd = s0 + r + 8 * hh;
      int vv  = ct * 16 + mr;
      kvout[kdd * DV + vv] = acc[ct][r];
    }
  }
}

// ---------------------------------------------------------------------------
// Pass 2: sequential inter-chunk scan (state in registers), emit prev-states
// ---------------------------------------------------------------------------
__global__ __launch_bounds__(256) void mlstm_p2(
    const float* __restrict__ C0, const float* __restrict__ n0,
    const float* __restrict__ m0,
    const float* __restrict__ ws_kv, const float* __restrict__ ws_ksum,
    const float* __restrict__ ws_lfl, const float* __restrict__ ws_mi,
    float* __restrict__ ws_cprev, float* __restrict__ ws_nprev,
    float* __restrict__ ws_mprev)
{
  __shared__ float s_m;
  const int tid = threadIdx.x;
  const int bh  = blockIdx.x;
  float cst[64];
  #pragma unroll
  for (int i = 0; i < 64; ++i)
    cst[i] = C0[(size_t)bh * (DK * DV) + tid + 256 * i];
  float nv = 0.f;
  if (tid < DK) nv = n0[(size_t)bh * DK + tid];
  if (tid == 0) s_m = m0[bh];
  __syncthreads();
  for (int j = 0; j < NCH; ++j) {
    const size_t blk = (size_t)bh * NCH + j;
    float* cp = ws_cprev + blk * (DK * DV);
    #pragma unroll
    for (int i = 0; i < 64; ++i) cp[tid + 256 * i] = cst[i];
    if (tid < DK) ws_nprev[blk * DK + tid] = nv;
    float mprev = s_m;
    if (tid == 0) ws_mprev[blk] = mprev;
    float lfl = ws_lfl[blk];
    float mi  = ws_mi[blk];
    float mn  = fmaxf(lfl + mprev, mi);
    float e1  = expf(lfl + mprev - mn);
    float e2  = expf(mi - mn);
    const float* kv = ws_kv + blk * (DK * DV);
    #pragma unroll
    for (int i = 0; i < 64; ++i)
      cst[i] = cst[i] * e1 + kv[tid + 256 * i] * e2;
    if (tid < DK) nv = nv * e1 + ws_ksum[blk * DK + tid] * e2;
    __syncthreads();
    if (tid == 0) s_m = mn;
    __syncthreads();
  }
}

// ---------------------------------------------------------------------------
// Pass 3: scores = q k^T (WMMA) -> G mask -> h = M v + (q*iw) c_prev (WMMA)
// ---------------------------------------------------------------------------
__global__ __launch_bounds__(256) void mlstm_p3(
    const float* __restrict__ qg, const float* __restrict__ kg,
    const float* __restrict__ vg, const float* __restrict__ ig,
    const float* __restrict__ ws_lfacc, const float* __restrict__ ws_pm,
    const float* __restrict__ ws_cprev, const float* __restrict__ ws_nprev,
    const float* __restrict__ ws_mprev,
    float* __restrict__ outg)
{
  extern __shared__ char smem[];
  unsigned short* QB  = (unsigned short*)smem;    // [s][k] (q*scale; later *iw)
  unsigned short* KB  = QB + SCH * SB;            // [t][k]
  unsigned short* VT  = KB + SCH * SB;            // [v][t]
  unsigned short* MB  = VT + SCH * SB;            // [s][t]
  unsigned short* CBT = MB + SCH * SB;            // [v][k]  (c_prev^T, bf16)
  float* lfa  = (float*)(CBT + SCH * SB);
  float* ivv  = lfa + SCH;
  float* pmv  = ivv + SCH;
  float* gst  = pmv + SCH;
  float* iwv  = gst + SCH;
  float* eng  = iwv + SCH;
  float* rsum = eng + SCH;
  float* rinv = rsum + SCH;
  float* itn  = rinv + SCH;
  float* nstl = itn + SCH;
  float* scl  = nstl + SCH;

  const int tid = threadIdx.x;
  const int blk = blockIdx.x;
  const int bh  = blk >> 5;
  const int j   = blk & 31;
  const int w = tid >> 5, ln = tid & 31, mr = ln & 15, hh = ln >> 4;
  const int s0 = w * 16;
  const float qscale = 0.08838834764831845f;      // 1/sqrt(128)

  const float* qc = qg + ((size_t)bh * TSEQ + (size_t)j * SCH) * DK;
  const float* kc = kg + ((size_t)bh * TSEQ + (size_t)j * SCH) * DK;
  const float* vc = vg + ((size_t)bh * TSEQ + (size_t)j * SCH) * DV;
  const float* ic = ig + (size_t)bh * TSEQ + (size_t)j * SCH;
  float* op = outg + ((size_t)bh * TSEQ + (size_t)j * SCH) * DV;

  for (int e = tid; e < SCH * DK / 4; e += 256) {
    int row = e >> 5;
    int c4  = (e & 31) << 2;
    float4 xq = *reinterpret_cast<const float4*>(qc + row * DK + c4);
    QB[row * SB + c4 + 0] = f2bfu(xq.x * qscale);
    QB[row * SB + c4 + 1] = f2bfu(xq.y * qscale);
    QB[row * SB + c4 + 2] = f2bfu(xq.z * qscale);
    QB[row * SB + c4 + 3] = f2bfu(xq.w * qscale);
    float4 xk = *reinterpret_cast<const float4*>(kc + row * DK + c4);
    KB[row * SB + c4 + 0] = f2bfu(xk.x);
    KB[row * SB + c4 + 1] = f2bfu(xk.y);
    KB[row * SB + c4 + 2] = f2bfu(xk.z);
    KB[row * SB + c4 + 3] = f2bfu(xk.w);
    float4 xv = *reinterpret_cast<const float4*>(vc + row * DV + c4);
    VT[(c4 + 0) * SB + row] = f2bfu(xv.x);
    VT[(c4 + 1) * SB + row] = f2bfu(xv.y);
    VT[(c4 + 2) * SB + row] = f2bfu(xv.z);
    VT[(c4 + 3) * SB + row] = f2bfu(xv.w);
  }
  const float* cp = ws_cprev + (size_t)blk * (DK * DV);
  for (int e = tid; e < DK * DV; e += 256) {
    int kk = e >> 7, vv = e & 127;
    CBT[vv * SB + kk] = f2bfu(cp[e]);
  }
  if (tid < SCH) {
    lfa[tid]  = ws_lfacc[(size_t)blk * SCH + tid];
    pmv[tid]  = ws_pm[(size_t)blk * SCH + tid];
    ivv[tid]  = ic[tid];
    nstl[tid] = ws_nprev[(size_t)blk * DK + tid];
    rsum[tid] = 0.f;
  }
  if (tid == 0) scl[0] = ws_mprev[blk];
  __syncthreads();
  if (tid < SCH) {
    int s = tid;
    float mp = scl[0];
    float gs = lfa[s] + fmaxf(pmv[s], mp);   // G_stab
    gst[s] = gs;
    iwv[s] = expf(lfa[s] + mp - gs);         // inter_w
    eng[s] = expf(-gs);
    float dot = 0.f;
    for (int kk = 0; kk < DK; ++kk) dot += bfu2f(QB[s * SB + kk]) * nstl[kk];
    itn[s] = dot * iwv[s];                   // inter_n
  }
  __syncthreads();

  // matmul1: scores = q k^T, then mask by G, write M (bf16) + row sums
  {
    v8f acc[8];
    v8f z8 = {0.f, 0.f, 0.f, 0.f, 0.f, 0.f, 0.f, 0.f};
    #pragma unroll
    for (int ct = 0; ct < 8; ++ct) acc[ct] = z8;
    #pragma unroll
    for (int kt = 0; kt < 4; ++kt) {
      const int kb16 = kt * 32;
      v16bf A = load_a(QB, s0, mr, hh, kb16);
      #pragma unroll
      for (int ct = 0; ct < 8; ++ct) {
        v16bf Bv = load_b(KB, ct, mr, hh, kb16);
        acc[ct] = __builtin_amdgcn_wmma_f32_16x16x32_bf16(
            false, A, false, Bv, (short)0, acc[ct], false, false);
      }
    }
    float part[8] = {0.f, 0.f, 0.f, 0.f, 0.f, 0.f, 0.f, 0.f};
    #pragma unroll
    for (int ct = 0; ct < 8; ++ct) {
      #pragma unroll
      for (int r = 0; r < 8; ++r) {
        int s = s0 + r + 8 * hh;
        int t = ct * 16 + mr;
        float g   = (t <= s) ? expf(lfa[s] - lfa[t] + ivv[t] - gst[s]) : 0.f;
        float val = acc[ct][r] * g;
        MB[s * SB + t] = f2bfu(val);
        part[r] += val;
      }
    }
    #pragma unroll
    for (int r = 0; r < 8; ++r) atomicAdd(&rsum[s0 + r + 8 * hh], part[r]);
  }
  __syncthreads();
  if (tid < SCH) {
    float Mn = fmaxf(fabsf(rsum[tid] + itn[tid]), eng[tid]);
    rinv[tid] = 1.f / Mn;
  }
  // Rescale QB rows in place by inter_w[s] (q no longer needed unscaled).
  for (int e = tid; e < SCH * DK; e += 256) {
    int row = e >> 7;
    int col = e & 127;
    QB[row * SB + col] = f2bfu(bfu2f(QB[row * SB + col]) * iwv[row]);
  }
  __syncthreads();

  // matmul2 + matmul3: h = M*v + (q*inter_w)*c_prev, then scale rows by rinv
  {
    v8f acc[8];
    v8f z8 = {0.f, 0.f, 0.f, 0.f, 0.f, 0.f, 0.f, 0.f};
    #pragma unroll
    for (int ct = 0; ct < 8; ++ct) acc[ct] = z8;
    #pragma unroll
    for (int kt = 0; kt < 4; ++kt) {
      const int kb16 = kt * 32;
      v16bf A = load_a(MB, s0, mr, hh, kb16);
      #pragma unroll
      for (int ct = 0; ct < 8; ++ct) {
        v16bf Bv = load_b(VT, ct, mr, hh, kb16);
        acc[ct] = __builtin_amdgcn_wmma_f32_16x16x32_bf16(
            false, A, false, Bv, (short)0, acc[ct], false, false);
      }
    }
    #pragma unroll
    for (int kt = 0; kt < 4; ++kt) {
      const int kb16 = kt * 32;
      v16bf A = load_a(QB, s0, mr, hh, kb16);
      #pragma unroll
      for (int ct = 0; ct < 8; ++ct) {
        v16bf Bv = load_b(CBT, ct, mr, hh, kb16);
        acc[ct] = __builtin_amdgcn_wmma_f32_16x16x32_bf16(
            false, A, false, Bv, (short)0, acc[ct], false, false);
      }
    }
    #pragma unroll
    for (int ct = 0; ct < 8; ++ct) {
      #pragma unroll
      for (int r = 0; r < 8; ++r) {
        int s  = s0 + r + 8 * hh;
        int vv = ct * 16 + mr;
        op[(size_t)s * DV + vv] = acc[ct][r] * rinv[s];
      }
    }
  }
}

// ---------------------------------------------------------------------------
extern "C" void kernel_launch(void* const* d_in, const int* in_sizes, int n_in,
                              void* d_out, int out_size, void* d_ws, size_t ws_size,
                              hipStream_t stream) {
  (void)in_sizes; (void)n_in; (void)out_size; (void)ws_size;
  const float* q  = (const float*)d_in[0];
  const float* k  = (const float*)d_in[1];
  const float* v  = (const float*)d_in[2];
  const float* ig = (const float*)d_in[3];
  const float* fg = (const float*)d_in[4];
  const float* C0 = (const float*)d_in[5];
  const float* n0 = (const float*)d_in[6];
  const float* m0 = (const float*)d_in[7];
  float* out = (float*)d_out;

  float* W = (float*)d_ws;
  const size_t NB = (size_t)NBH * NCH;   // 512 (head,chunk) tiles
  float* ws_kv    = W;                          // 512*16384
  float* ws_cprev = ws_kv    + NB * DK * DV;    // 512*16384
  float* ws_ksum  = ws_cprev + NB * DK * DV;    // 512*128
  float* ws_nprev = ws_ksum  + NB * DK;         // 512*128
  float* ws_lfacc = ws_nprev + NB * DK;         // 512*128
  float* ws_pm    = ws_lfacc + NB * SCH;        // 512*128
  float* ws_lfl   = ws_pm    + NB * SCH;        // 512
  float* ws_mi    = ws_lfl   + NB;              // 512
  float* ws_mprev = ws_mi    + NB;              // 512

  constexpr int SMEM1 = 3 * SCH * SB * 2 + (4 * SCH + 8) * 4;   // ~103.5 KB
  constexpr int SMEM3 = 5 * SCH * SB * 2 + (10 * SCH + 8) * 4;  // ~174 KB
  hipFuncSetAttribute(reinterpret_cast<const void*>(mlstm_p1),
                      hipFuncAttributeMaxDynamicSharedMemorySize, SMEM1);
  hipFuncSetAttribute(reinterpret_cast<const void*>(mlstm_p3),
                      hipFuncAttributeMaxDynamicSharedMemorySize, SMEM3);

  mlstm_p1<<<dim3((unsigned)NB), dim3(256), SMEM1, stream>>>(
      k, v, ig, fg, ws_kv, ws_ksum, ws_lfacc, ws_pm, ws_lfl, ws_mi);
  mlstm_p2<<<dim3(NBH), dim3(256), 0, stream>>>(
      C0, n0, m0, ws_kv, ws_ksum, ws_lfl, ws_mi, ws_cprev, ws_nprev, ws_mprev);
  mlstm_p3<<<dim3((unsigned)NB), dim3(256), SMEM3, stream>>>(
      q, k, v, ig, ws_lfacc, ws_pm, ws_cprev, ws_nprev, ws_mprev, out);
}
